// WLNet_12816182411667
// MI455X (gfx1250) — matrix operand order repaired and checked by reading, not compiled, not run on hardware
//
#include <hip/hip_runtime.h>

typedef __attribute__((ext_vector_type(16))) __bf16 v16bf;
typedef __attribute__((ext_vector_type(8)))  float  v8f;
typedef __attribute__((ext_vector_type(4)))  unsigned v4u;
typedef __attribute__((ext_vector_type(8)))  int      v8i;
typedef __attribute__((ext_vector_type(4)))  int      v4i;

#define NNODE 1024
#define NEDGE 16384
#define NPOS  8192
#define DD    64
#define LNEPS 1e-5f

// ---------------------------------------------------------------- utilities
__global__ void k_zero_f(float* p, int n) {
  int t = blockIdx.x * blockDim.x + threadIdx.x;
  if (t < n) p[t] = 0.0f;
}
__global__ void k_zero_u(unsigned* p, int n) {
  int t = blockIdx.x * blockDim.x + threadIdx.x;
  if (t < n) p[t] = 0u;
}

// h = emb[x]
__global__ void k_embed(const int* __restrict__ x, const float* __restrict__ emb,
                        float* __restrict__ h) {
  int t = blockIdx.x * blockDim.x + threadIdx.x;
  if (t < NNODE * DD) {
    int node = t >> 6, d = t & 63;
    h[t] = emb[x[node] * DD + d];
  }
}

// degree + adjacency bitset
__global__ void k_edges(const int* __restrict__ ei, float* __restrict__ deg,
                        unsigned* __restrict__ adj) {
  int e = blockIdx.x * blockDim.x + threadIdx.x;
  if (e < NEDGE) {
    int s = ei[e], d = ei[NEDGE + e];
    atomicAdd(&deg[d], 1.0f);
    unsigned b = (unsigned)s * NNODE + (unsigned)d;
    atomicOr(&adj[b >> 5], 1u << (b & 31));
  }
}

// aggr[dst] += h[src]
__global__ void k_aggr(const int* __restrict__ ei, const float* __restrict__ h,
                       float* __restrict__ aggr) {
  int t = blockIdx.x * blockDim.x + threadIdx.x;
  if (t < NEDGE * DD) {
    int e = t >> 6, d = t & 63;
    int s = ei[e], dst = ei[NEDGE + e];
    atomicAdd(&aggr[dst * DD + d], h[s * DD + d]);
  }
}

// per-node: LN(aggr/deg @ Wl + bl + h @ Wr).  one block (64 threads) per node
__global__ __launch_bounds__(64) void k_sage(
    const float* __restrict__ aggr, const float* __restrict__ deg,
    const float* __restrict__ hcur, const float* __restrict__ Wl,
    const float* __restrict__ bl, const float* __restrict__ Wr,
    float* __restrict__ hnext) {
  __shared__ float sa[DD], sh[DD], red[DD];
  int node = blockIdx.x, d = threadIdx.x;
  float dg = fmaxf(deg[node], 1.0f);
  sa[d] = aggr[node * DD + d] / dg;
  sh[d] = hcur[node * DD + d];
  __syncthreads();
  float y = bl[d];
  #pragma unroll 8
  for (int k = 0; k < DD; ++k)
    y += sa[k] * Wl[k * DD + d] + sh[k] * Wr[k * DD + d];
  red[d] = y;
  __syncthreads();
  for (int off = 32; off >= 1; off >>= 1) {
    if (d < off) red[d] += red[d + off];
    __syncthreads();
  }
  float mean = red[0] * (1.0f / DD);
  __syncthreads();
  float c = y - mean;
  red[d] = c * c;
  __syncthreads();
  for (int off = 32; off >= 1; off >>= 1) {
    if (d < off) red[d] += red[d + off];
    __syncthreads();
  }
  float inv = rsqrtf(red[0] * (1.0f / DD) + LNEPS);
  hnext[node * DD + d] = c * inv;
}

// ------------------------------------------------------------ helpers
static __device__ __forceinline__ float adj_bit(const unsigned* adj, int i, int j) {
  unsigned b = (unsigned)i * NNODE + (unsigned)j;
  return ((adj[b >> 5] >> (b & 31)) & 1u) ? 1.0f : 0.0f;
}

// TDM: DMA a [rows x 64] f32 tile (row stride 64) from global into LDS.
// D# group0: count=1, lds_addr, 57-bit global addr, type=2.
// D# group1: data_size=4B, tensor_dim0=64, tensor_dim1=1024, tile 64 x rows,
//            tensor_dim0_stride=64.  Groups 2/3 zero (2D tile).
static __device__ __forceinline__ void tdm_load_rows64(unsigned lds_off,
                                                       const float* gptr,
                                                       int rows) {
  unsigned long long ga = (unsigned long long)(uintptr_t)gptr;
  v4u g0;
  g0[0] = 1u;                                   // count=1 (user descriptor)
  g0[1] = lds_off;                              // lds_addr
  g0[2] = (unsigned)ga;                         // global_addr[31:0]
  g0[3] = (unsigned)(ga >> 32) | 0x80000000u;   // global_addr[56:32] | type=2
  v8i g1;
  g1[0] = 0x00020000;                           // data_size = 4 bytes
  g1[1] = (64 << 16);                           // tensor_dim0 = 64
  g1[2] = (NNODE << 16);                        // tensor_dim1 = 1024
  g1[3] = (64 << 16);                           // tile_dim0 = 64
  g1[4] = rows;                                 // tile_dim1 = rows
  g1[5] = 64;                                   // tensor_dim0_stride = 64
  g1[6] = 0;
  g1[7] = 0;
  v4i z4 = {0, 0, 0, 0};
  v8i z8 = {0, 0, 0, 0, 0, 0, 0, 0};
  __builtin_amdgcn_tensor_load_to_lds(g0, g1, z4, z4, z8, 0);
}

// ---------------------------------------------------------------- B-pass
// grid (64,64): tile of 16 i x 16 j pairs.  M = 256 pair-rows, K=64, N=64.
// B = relu(LN(A @ h_W + h_b)); accumulate row sums (vec_p) / col sums (vec_q).
__global__ __launch_bounds__(256) void k_bsum(
    const float* __restrict__ h, const float* __restrict__ hW,
    const float* __restrict__ hb, const unsigned* __restrict__ adj,
    float* __restrict__ vp, float* __restrict__ vq) {
  __shared__ __bf16 sA[256][72];    // pair rows, bf16, padded stride
  __shared__ float  sHi[16][DD];
  __shared__ float  sHj[16][DD];
  __shared__ float  sQ[16][DD];     // vec_q partials across the 16 i's
  int tid = threadIdx.x;
  int lane = tid & 31, wave = tid >> 5;
  int i0 = blockIdx.y * 16, j0 = blockIdx.x * 16;

  __builtin_prefetch(hW, 0, 0);     // global_prefetch_b8: weights

  // TDM async-tensor DMA of the two 16x64 h tiles into LDS (wave 0 issues;
  // TDM ignores EXEC so one wave -> one DMA each; TENSORcnt is per-wave).
  if (wave == 0) {
    tdm_load_rows64((unsigned)(uintptr_t)&sHi[0][0], h + i0 * DD, 16);
    tdm_load_rows64((unsigned)(uintptr_t)&sHj[0][0], h + j0 * DD, 16);
  }
  for (int t = tid; t < 16 * DD; t += 256) sQ[t >> 6][t & 63] = 0.0f;

  // preload h_W B-fragments in registers: 2 k-steps x 4 n-tiles (overlaps DMA)
  int n = lane & 15, g = lane >> 4;
  v16bf bfr[2][4];
  #pragma unroll
  for (int kk = 0; kk < 2; ++kk)
    #pragma unroll
    for (int a = 0; a < 4; ++a)
      #pragma unroll
      for (int e = 0; e < 16; ++e) {
        int kg = kk * 32 + g * 16 + e;             // B layout: K = 16*g + e
        bfr[kk][a][e] = (__bf16)hW[kg * DD + a * 16 + n];
      }

  if (wave == 0) __builtin_amdgcn_s_wait_tensorcnt(0);
  __syncthreads();                                 // publish TDM'd LDS data

  // thread p builds pair-row p:  row(k) = h[i0+p/16][k] * h[j0+p%16][k]
  {
    int ii = tid >> 4, jj = tid & 15;
    #pragma unroll 8
    for (int k = 0; k < DD; ++k)
      sA[tid][k] = (__bf16)(sHi[ii][k] * sHj[jj][k]);
  }
  __syncthreads();

  float vqacc[8][4];
  #pragma unroll
  for (int r = 0; r < 8; ++r)
    #pragma unroll
    for (int a = 0; a < 4; ++a) vqacc[r][a] = 0.0f;

  const v8f vzero = {0.f, 0.f, 0.f, 0.f, 0.f, 0.f, 0.f, 0.f};
  for (int tt = 0; tt < 2; ++tt) {                 // 2 M-tiles per wave
    int t = wave + tt * 8;
    int i = i0 + t;                                // fixed i for this M-tile
    v8f acc[4] = {vzero, vzero, vzero, vzero};
    #pragma unroll
    for (int kk = 0; kk < 2; ++kk) {
      v16bf af;                                    // A layout: M = lane&15
      #pragma unroll
      for (int e = 0; e < 16; ++e) {
        int k = kk * 32 + g * 16 + e;
        af[e] = sA[t * 16 + n][k];
      }
      #pragma unroll
      for (int a = 0; a < 4; ++a)
        acc[a] = __builtin_amdgcn_wmma_f32_16x16x32_bf16(
            false, af, false, bfr[kk][a], (short)0, acc[a], false, false);
    }
    // epilogue: +bias +edge-term, LayerNorm over 64, relu, accumulate sums
    float vpacc[4] = {0.f, 0.f, 0.f, 0.f};
    #pragma unroll
    for (int r = 0; r < 8; ++r) {
      int mrow = g * 8 + r;                        // C layout: M = 8g + r
      int j = j0 + mrow;
      float eb = adj_bit(adj, i, j);
      float y[4], s1 = 0.f, s2 = 0.f;
      #pragma unroll
      for (int a = 0; a < 4; ++a) {
        y[a] = acc[a][r] + hb[a * 16 + n] + eb * hW[64 * DD + a * 16 + n];
        s1 += y[a];
        s2 += y[a] * y[a];
      }
      #pragma unroll
      for (int off = 1; off < 16; off <<= 1) {     // reduce across 16 lanes
        s1 += __shfl_xor(s1, off, 32);
        s2 += __shfl_xor(s2, off, 32);
      }
      float mean = s1 * (1.0f / DD);
      float inv = rsqrtf(s2 * (1.0f / DD) - mean * mean + LNEPS);
      #pragma unroll
      for (int a = 0; a < 4; ++a) {
        float v = fmaxf((y[a] - mean) * inv, 0.0f);
        vpacc[a] += v;            // sum over j (this lane's d)
        vqacc[r][a] += v;         // sum over i (register acc across tt)
      }
    }
    #pragma unroll
    for (int a = 0; a < 4; ++a) {                  // flush vec_p for this i
      float full = vpacc[a] + __shfl_xor(vpacc[a], 16, 32);
      if (g == 0) atomicAdd(&vp[i * DD + a * 16 + n], full);
    }
  }
  // vec_q: LDS reduce across waves, then one global flush
  #pragma unroll
  for (int r = 0; r < 8; ++r)
    #pragma unroll
    for (int a = 0; a < 4; ++a)
      atomicAdd(&sQ[g * 8 + r][a * 16 + n], vqacc[r][a]);
  __syncthreads();
  for (int t = tid; t < 16 * DD; t += 256) {
    int m = t >> 6, d = t & 63;
    atomicAdd(&vq[(j0 + m) * DD + d], sQ[m][d]);
  }
}

// ---------------------------------------------------------------- G-pass
// 16384 selected rows ((a,b) then (b,a)).  M=64/WG, K=193 pad 224, N=64.
__global__ __launch_bounds__(128) void k_gsel(
    const float* __restrict__ h, const unsigned* __restrict__ adj,
    const float* __restrict__ vp, const float* __restrict__ vq,
    const int* __restrict__ pos, const float* __restrict__ gW,
    const float* __restrict__ gb, float* __restrict__ Gsel) {
  __shared__ __bf16 sA[64][232];
  __shared__ __bf16 sB[224][72];
  int tid = threadIdx.x, lane = tid & 31, wave = tid >> 5;
  int base = blockIdx.x * 64;

  for (int t = tid; t < 224 * DD; t += 128) {
    int k = t >> 6, c = t & 63;
    sB[k][c] = (__bf16)((k < 193) ? gW[k * DD + c] : 0.0f);
  }
  for (int t = tid; t < 64 * 224; t += 128) {
    int r = t / 224, k = t % 224;
    int ridx = base + r;
    int pr = ridx & (NPOS - 1), sw = ridx >> 13;
    int i = pos[pr * 2 + sw];
    int j = pos[pr * 2 + (sw ^ 1)];
    float v;
    if (k < 64)       v = h[i * DD + k] * h[j * DD + k];
    else if (k == 64) v = adj_bit(adj, i, j);
    else if (k < 129) v = vp[i * DD + (k - 65)];
    else if (k < 193) v = vq[j * DD + (k - 129)];
    else              v = 0.0f;
    sA[r][k] = (__bf16)v;
  }
  __syncthreads();

  int n = lane & 15, g = lane >> 4;
  const v8f vzero = {0.f, 0.f, 0.f, 0.f, 0.f, 0.f, 0.f, 0.f};
  v8f acc[4] = {vzero, vzero, vzero, vzero};
  #pragma unroll
  for (int kk = 0; kk < 7; ++kk) {
    v16bf af;
    #pragma unroll
    for (int e = 0; e < 16; ++e) {
      int k = kk * 32 + g * 16 + e;
      af[e] = sA[wave * 16 + n][k];
    }
    #pragma unroll
    for (int a = 0; a < 4; ++a) {
      v16bf bfrag;
      #pragma unroll
      for (int e = 0; e < 16; ++e)
        bfrag[e] = sB[kk * 32 + g * 16 + e][a * 16 + n];
      acc[a] = __builtin_amdgcn_wmma_f32_16x16x32_bf16(
          false, af, false, bfrag, (short)0, acc[a], false, false);
    }
  }
  #pragma unroll
  for (int r = 0; r < 8; ++r) {
    int m = g * 8 + r;
    int row = base + wave * 16 + m;
    float y[4], s1 = 0.f, s2 = 0.f;
    #pragma unroll
    for (int a = 0; a < 4; ++a) {
      y[a] = acc[a][r] + gb[a * 16 + n];
      s1 += y[a];
      s2 += y[a] * y[a];
    }
    #pragma unroll
    for (int off = 1; off < 16; off <<= 1) {
      s1 += __shfl_xor(s1, off, 32);
      s2 += __shfl_xor(s2, off, 32);
    }
    float mean = s1 * (1.0f / DD);
    float inv = rsqrtf(s2 * (1.0f / DD) - mean * mean + LNEPS);
    #pragma unroll
    for (int a = 0; a < 4; ++a)
      Gsel[row * DD + a * 16 + n] = fmaxf((y[a] - mean) * inv, 0.0f);
  }
}

// out[p] = sum_d G[a,b,d]*G[b,a,d]*linW[d] + linb
__global__ void k_out(const float* __restrict__ Gsel,
                      const float* __restrict__ linW,
                      const float* __restrict__ linb, float* __restrict__ out) {
  int p = blockIdx.x * blockDim.x + threadIdx.x;
  if (p < NPOS) {
    float s = 0.0f;
    #pragma unroll 8
    for (int d = 0; d < DD; ++d)
      s += Gsel[p * DD + d] * Gsel[(p + NPOS) * DD + d] * linW[d];
    out[p] = s + linb[0];
  }
}

// ---------------------------------------------------------------- launcher
extern "C" void kernel_launch(void* const* d_in, const int* in_sizes, int n_in,
                              void* d_out, int out_size, void* d_ws, size_t ws_size,
                              hipStream_t stream) {
  const int*   x    = (const int*)d_in[0];
  const int*   ei   = (const int*)d_in[1];
  const int*   pos  = (const int*)d_in[2];
  const float* emb  = (const float*)d_in[3];
  const float* Wl   = (const float*)d_in[4];
  const float* bl   = (const float*)d_in[5];
  const float* Wr   = (const float*)d_in[6];
  const float* hW   = (const float*)d_in[7];
  const float* hb   = (const float*)d_in[8];
  const float* gW   = (const float*)d_in[9];
  const float* gb   = (const float*)d_in[10];
  const float* linW = (const float*)d_in[11];
  const float* linb = (const float*)d_in[12];

  float* ws   = (float*)d_ws;
  float* hA   = ws;                    // 65536
  float* hB   = ws + 65536;            // 65536
  float* aggr = ws + 131072;           // 65536
  float* deg  = ws + 196608;           // 1024
  float* vp   = ws + 197632;           // 65536
  float* vq   = ws + 263168;           // 65536
  float* Gsel = ws + 328704;           // 16384*64
  unsigned* adj = (unsigned*)(ws + 328704 + 16384 * 64);  // 32768 u32

  // zero accumulators (aggr, deg, vec_p, vec_q are contiguous) + adjacency
  k_zero_f<<<(197632 + 255) / 256, 256, 0, stream>>>(aggr, 197632);
  k_zero_u<<<(32768 + 255) / 256, 256, 0, stream>>>(adj, 32768);
  k_embed<<<(NNODE * DD) / 256, 256, 0, stream>>>(x, emb, hA);
  k_edges<<<NEDGE / 256, 256, 0, stream>>>(ei, deg, adj);

  for (int layer = 0; layer < 2; ++layer) {
    if (layer)
      k_zero_f<<<(NNODE * DD) / 256, 256, 0, stream>>>(aggr, NNODE * DD);
    const float* cur = layer ? hB : hA;
    float* nxt = layer ? hA : hB;
    k_aggr<<<(NEDGE * DD) / 256, 256, 0, stream>>>(ei, cur, aggr);
    k_sage<<<NNODE, DD, 0, stream>>>(aggr, deg, cur,
                                     Wl + layer * DD * DD, bl + layer * DD,
                                     Wr + layer * DD * DD, nxt);
  }
  // final h lives in hA
  k_bsum<<<dim3(64, 64), 256, 0, stream>>>(hA, hW, hb, adj, vp, vq);
  k_gsel<<<256, 128, 0, stream>>>(hA, adj, vp, vq, pos, gW, gb, Gsel);
  k_out<<<NPOS / 256, 256, 0, stream>>>(Gsel, linW, linb, (float*)d_out);
}
